// MultiHeadLatentAttention_13975823581581
// MI455X (gfx1250) — matrix operand order repaired and checked by reading, not compiled, not run on hardware
//
#include <hip/hip_runtime.h>
#include <stdint.h>

// ---------------------------------------------------------------------------
// MLA forward for MI455X (gfx1250, wave32, WMMA + TDM).
// Heavy matrix math: v_wmma_f32_16x16x32_bf16 (fp32 accumulate).
// One-time f32->bf16 conversion/transpose pre-passes keep the GEMM hot loop
// conversion-free; tiles double-buffered into LDS by the Tensor Data Mover
// (tensor_load_to_lds + s_wait_tensorcnt) so DMA overlaps WMMA.
// ---------------------------------------------------------------------------

typedef __attribute__((ext_vector_type(16))) __bf16 bf16x16;
typedef __attribute__((ext_vector_type(8)))  __bf16 bf16x8;
typedef __attribute__((ext_vector_type(8)))  float  f32x8;

#define DIMSZ  2048
#define NH     16
#define HDSZ   128
#define SEQ    2048
#define NB     2
#define LATD   512
#define W3     3072   // DIM + ROPE*H == NOPE*H + DIM

#if defined(__AMDGCN__) && defined(__has_builtin)
#  if __has_builtin(__builtin_amdgcn_tensor_load_to_lds)
#    define HAVE_TDM 1
#  endif
#endif
#ifndef HAVE_TDM
#  define HAVE_TDM 0
#endif

__device__ __forceinline__ unsigned short f32_bf16(float f) {
  union { float f; unsigned int u; } c; c.f = f;
  unsigned int u = c.u;
  u += 0x7fffu + ((u >> 16) & 1u);   // round-to-nearest-even
  return (unsigned short)(u >> 16);
}

__device__ __forceinline__ f32x8 splat8(float x) {
  f32x8 r;
  #pragma unroll
  for (int i = 0; i < 8; ++i) r[i] = x;
  return r;
}

// A-matrix fragment: two 16B chunks; call sites pass adjacent ones (k = 16*hi + s, as in B).
__device__ __forceinline__ bf16x16 frag_from2(const unsigned short* p0,
                                              const unsigned short* p1) {
  bf16x8 a = *(const bf16x8*)p0;
  bf16x8 b = *(const bf16x8*)p1;
  return __builtin_shufflevector(a, b, 0,1,2,3,4,5,6,7,8,9,10,11,12,13,14,15);
}

#if HAVE_TDM
// Issue one 2D TDM tile load: `tile_rows` rows of `tile_k` bf16 elements,
// row stride `stride_elems`, into LDS at byte offset lds_off (rows packed
// contiguously, tile_k*2 bytes per row). D# per CDNA5 ISA ch.8.
// This toolchain (clang-23 / therock-10.0): 6-arg builtin
//   (uint32x4 g0, int32x8 g1, int32x4 g2, int32x4 g3, int32x8 g4, i32 cpol)
__device__ __forceinline__ void tdm_load_tile_2d(
    const void* gptr, unsigned int lds_off,
    unsigned int dim0_elems, unsigned int dim1_rows,
    unsigned int tile_k, unsigned int tile_rows,
    unsigned long long stride_elems)
{
  typedef __attribute__((ext_vector_type(4))) unsigned int u32x4;
  typedef __attribute__((ext_vector_type(8))) int i32x8;
  typedef __attribute__((ext_vector_type(4))) int i32x4;

  unsigned long long ga = (unsigned long long)(uintptr_t)gptr;
  u32x4 g0;
  g0[0] = 1u;                                   // count=1, user descriptor
  g0[1] = lds_off;                              // lds_addr (bytes)
  g0[2] = (unsigned int)ga;                     // global_addr[31:0]
  g0[3] = (unsigned int)(ga >> 32) | (2u << 30);// global_addr[56:32] | type=2

  i32x8 g1;
  g1[0] = (int)(1u << 16);                      // data_size=1 (2 bytes)
  g1[1] = (int)((dim0_elems & 0xFFFFu) << 16);  // tensor_dim0[15:0] @ bits63:48
  g1[2] = (int)(((dim0_elems >> 16) & 0xFFFFu) |
                ((dim1_rows & 0xFFFFu) << 16)); // dim0[31:16] | dim1[15:0]
  g1[3] = (int)(((dim1_rows >> 16) & 0xFFFFu) |
                ((tile_k & 0xFFFFu) << 16));    // dim1[31:16] | tile_dim0
  g1[4] = (int)(tile_rows & 0xFFFFu);           // tile_dim1 (tile_dim2=0)
  g1[5] = (int)(unsigned int)(stride_elems & 0xFFFFFFFFull);   // dim0_stride lo
  g1[6] = (int)(unsigned int)((stride_elems >> 32) & 0xFFFFull);// stride hi16
  g1[7] = 0;
  i32x4 z4 = {0, 0, 0, 0};
  i32x8 z8 = {0, 0, 0, 0, 0, 0, 0, 0};
  __builtin_amdgcn_tensor_load_to_lds(g0, g1, z4, z4, z8, 0);
}
#endif

// ---------------------------------------------------------------------------
// GEMM: C[M,N] f32 = A[M,K]bf16 * B^T (B stored [N,K] bf16, K-contiguous).
// Block 256 = 8 waves; tile 128(M) x 128(N) x 32(K); wave grid 4(M) x 2(N).
// Double-buffered TDM pipeline: wave 0 issues the NEXT K-tile pair into the
// alternate buffer, then waits TENSORcnt<=2 (in-order completion guarantees
// the CURRENT pair is resident); WMMA overlaps the in-flight DMA.
// ---------------------------------------------------------------------------
__global__ __launch_bounds__(256) void gemm_bf16_wmma(
    const unsigned short* __restrict__ A, const unsigned short* __restrict__ B,
    float* __restrict__ C, int M, int N, int K)
{
  __shared__ __align__(128) unsigned short a_lds[2][128 * 32]; // [m][k]
  __shared__ __align__(128) unsigned short b_lds[2][128 * 32]; // [n][k]

  const int t    = threadIdx.x;
  const int lane = t & 31;
  const int wave = t >> 5;
  const int hi   = lane >> 4;
  const int ln   = lane & 15;
  const int wm   = wave & 3;
  const int wn   = wave >> 2;
  const int m0   = blockIdx.y * 128;
  const int n0   = blockIdx.x * 128;

  f32x8 acc[2][4];
  #pragma unroll
  for (int i = 0; i < 2; ++i)
    #pragma unroll
    for (int j = 0; j < 4; ++j) acc[i][j] = splat8(0.f);

  const int nsteps = K >> 5;   // K/32

#if HAVE_TDM
  if (wave == 0) {             // preload step 0 into buffer 0
    tdm_load_tile_2d(A + (size_t)m0 * K,
                     (unsigned int)(uintptr_t)&a_lds[0][0],
                     (unsigned int)K, (unsigned int)M, 32u, 128u,
                     (unsigned long long)K);
    tdm_load_tile_2d(B + (size_t)n0 * K,
                     (unsigned int)(uintptr_t)&b_lds[0][0],
                     (unsigned int)K, (unsigned int)N, 32u, 128u,
                     (unsigned long long)K);
  }
#endif

  for (int i = 0; i < nsteps; ++i) {
    const int p = i & 1;
#if HAVE_TDM
    if (wave == 0) {
      if (i + 1 < nsteps) {
        const int k1 = (i + 1) << 5;
        tdm_load_tile_2d(A + (size_t)m0 * K + k1,
                         (unsigned int)(uintptr_t)&a_lds[p ^ 1][0],
                         (unsigned int)K, (unsigned int)M, 32u, 128u,
                         (unsigned long long)K);
        tdm_load_tile_2d(B + (size_t)n0 * K + k1,
                         (unsigned int)(uintptr_t)&b_lds[p ^ 1][0],
                         (unsigned int)K, (unsigned int)N, 32u, 128u,
                         (unsigned long long)K);
        __builtin_amdgcn_s_wait_tensorcnt(2);  // current pair resident
      } else {
        __builtin_amdgcn_s_wait_tensorcnt(0);  // drain
      }
    }
    __syncthreads();
#else
    {
      // cooperative fallback: 256 threads x 8 halves (one 16B chunk each, at a 16-half stride)
      const int k0 = i << 5;
      const int arow = (t * 16) >> 5;
      const int acol = (t * 16) & 31;
      *(uint4*)&a_lds[p][t * 16] =
          *(const uint4*)(A + (size_t)(m0 + arow) * K + k0 + acol);
      *(uint4*)&b_lds[p][t * 16] =
          *(const uint4*)(B + (size_t)(n0 + arow) * K + k0 + acol);
    }
    __syncthreads();
#endif

    bf16x16 afr[2], bfr[4];
    #pragma unroll
    for (int mt = 0; mt < 2; ++mt) {
      const unsigned short* pp = a_lds[p] + (wm * 32 + mt * 16 + ln) * 32;
      afr[mt] = frag_from2(pp + hi * 16, pp + hi * 16 + 8);
    }
    #pragma unroll
    for (int nt = 0; nt < 4; ++nt) {
      const unsigned short* pp = b_lds[p] + (wn * 64 + nt * 16 + ln) * 32 + hi * 16;
      bfr[nt] = *(const bf16x16*)pp;
    }
    #pragma unroll
    for (int mt = 0; mt < 2; ++mt)
      #pragma unroll
      for (int nt = 0; nt < 4; ++nt)
        acc[mt][nt] = __builtin_amdgcn_wmma_f32_16x16x32_bf16(
            false, afr[mt], false, bfr[nt], (short)0, acc[mt][nt], false, false);
    __syncthreads();   // readers done before buffer p is refilled (iter i+2)
  }

  #pragma unroll
  for (int mt = 0; mt < 2; ++mt) {
    const int mr = m0 + wm * 32 + mt * 16 + hi * 8;
    #pragma unroll
    for (int nt = 0; nt < 4; ++nt) {
      const int nc = n0 + wn * 64 + nt * 16 + ln;
      #pragma unroll
      for (int r = 0; r < 8; ++r)
        C[(size_t)(mr + r) * N + nc] = acc[mt][nt][r];
    }
  }
}

// ---------------------------------------------------------------------------
// One-time pre-passes.
// ---------------------------------------------------------------------------
__global__ __launch_bounds__(256) void cvt_f32_bf16(
    const float* __restrict__ in, unsigned short* __restrict__ out, int n4)
{
  int i = blockIdx.x * 256 + threadIdx.x;
  if (i >= n4) return;
  float4 f = *(const float4*)(in + (size_t)i * 4);
  unsigned int lo = (unsigned int)f32_bf16(f.x) | ((unsigned int)f32_bf16(f.y) << 16);
  unsigned int hiw = (unsigned int)f32_bf16(f.z) | ((unsigned int)f32_bf16(f.w) << 16);
  unsigned int* o = (unsigned int*)(out + (size_t)i * 4);
  o[0] = lo; o[1] = hiw;
}

// in: [R, C] f32 row-major -> out: [C, R] bf16 row-major
__global__ __launch_bounds__(256) void transpose_f32_bf16(
    const float* __restrict__ in, unsigned short* __restrict__ out, int R, int C)
{
  __shared__ float tile[32][33];
  const int c0 = blockIdx.x * 32, r0 = blockIdx.y * 32;
  const int tx = threadIdx.x & 31, ty = threadIdx.x >> 5;
  #pragma unroll
  for (int i = 0; i < 32; i += 8)
    tile[ty + i][tx] = in[(size_t)(r0 + ty + i) * C + c0 + tx];
  __syncthreads();
  #pragma unroll
  for (int i = 0; i < 32; i += 8)
    out[(size_t)(c0 + ty + i) * R + r0 + tx] = f32_bf16(tile[tx][ty + i]);
}

// RMS norm over rows of width n; f32 in -> bf16 out. One block per row.
__global__ __launch_bounds__(256) void rmsnorm_rows_bf16(
    const float* __restrict__ X, unsigned short* __restrict__ O, int n)
{
  const float* xr = X + (size_t)blockIdx.x * n;
  unsigned short* orow = O + (size_t)blockIdx.x * n;
  float ss = 0.f;
  for (int c = threadIdx.x; c < n; c += 256) { float v = xr[c]; ss += v * v; }
  #pragma unroll
  for (int m = 16; m; m >>= 1) ss += __shfl_xor(ss, m);
  __shared__ float red[8];
  __shared__ float rinv;
  if ((threadIdx.x & 31) == 0) red[threadIdx.x >> 5] = ss;
  __syncthreads();
  if (threadIdx.x == 0) {
    float s = 0.f;
    #pragma unroll
    for (int i = 0; i < 8; ++i) s += red[i];
    rinv = rsqrtf(s / (float)n + 1e-6f);
  }
  __syncthreads();
  for (int c = threadIdx.x; c < n; c += 256) orow[c] = f32_bf16(xr[c] * rinv);
}

// ---------------------------------------------------------------------------
// Per-(b,s) prep: segment RMS norms (64-wide), RoPE, q_gain; emit bf16
// q_full/k_full [B,H,S,128] and v transposed [B,H,128,S].
// ---------------------------------------------------------------------------
__global__ __launch_bounds__(256) void mla_prep(
    const float* __restrict__ q_krope,   // [B*S, 3072]
    const float* __restrict__ kv_up,     // [B*S, 3072]
    const float* __restrict__ q_gain,    // [16]
    unsigned short* __restrict__ q_full,
    unsigned short* __restrict__ k_full,
    unsigned short* __restrict__ v_t)
{
  const int s = blockIdx.x, b = blockIdx.y;
  const int t = threadIdx.x;
  const int h = t >> 4, sub = t & 15;
  const size_t row = (size_t)b * SEQ + s;
  const float* qr = q_krope + row * W3;
  const float* ur = kv_up   + row * W3;
  const size_t obase = (((size_t)(b * NH + h)) * SEQ + s) * HDSZ;
  const float g = q_gain[h];

  __shared__ float ls[NH][64];

  // q_nope
  {
    float v[4]; float ss = 0.f;
    #pragma unroll
    for (int j = 0; j < 4; ++j) { v[j] = qr[h * HDSZ + sub * 4 + j]; ss += v[j] * v[j]; }
    #pragma unroll
    for (int m = 8; m; m >>= 1) ss += __shfl_xor(ss, m);
    float r = rsqrtf(ss * (1.f / 64.f) + 1e-6f);
    #pragma unroll
    for (int j = 0; j < 4; ++j) q_full[obase + sub * 4 + j] = f32_bf16(v[j] * r * g);
  }
  // q_rope
  {
    float v[4]; float ss = 0.f;
    #pragma unroll
    for (int j = 0; j < 4; ++j) { v[j] = qr[h * HDSZ + 64 + sub * 4 + j]; ss += v[j] * v[j]; }
    #pragma unroll
    for (int m = 8; m; m >>= 1) ss += __shfl_xor(ss, m);
    float r = rsqrtf(ss * (1.f / 64.f) + 1e-6f);
    #pragma unroll
    for (int j = 0; j < 4; ++j) ls[h][sub * 4 + j] = v[j] * r;
  }
  __syncthreads();
  #pragma unroll
  for (int j = 0; j < 4; ++j) {
    int e = sub * 4 + j;
    int i = e & 31;
    float invf = __powf(10000.f, -(float)i / 32.f);
    float ang = (float)s * invf;
    float sn, c; __sincosf(ang, &sn, &c);
    float x1 = ls[h][i], x2 = ls[h][i + 32];
    float o = (e < 32) ? (x1 * c + x2 * sn) : (x2 * c - x1 * sn);
    q_full[obase + 64 + e] = f32_bf16(o * g);
  }
  __syncthreads();
  // k_nope
  {
    float v[4]; float ss = 0.f;
    #pragma unroll
    for (int j = 0; j < 4; ++j) { v[j] = ur[h * 64 + sub * 4 + j]; ss += v[j] * v[j]; }
    #pragma unroll
    for (int m = 8; m; m >>= 1) ss += __shfl_xor(ss, m);
    float r = rsqrtf(ss * (1.f / 64.f) + 1e-6f);
    #pragma unroll
    for (int j = 0; j < 4; ++j) k_full[obase + sub * 4 + j] = f32_bf16(v[j] * r);
  }
  // k_rope
  {
    float v[4]; float ss = 0.f;
    #pragma unroll
    for (int j = 0; j < 4; ++j) { v[j] = qr[DIMSZ + h * 64 + sub * 4 + j]; ss += v[j] * v[j]; }
    #pragma unroll
    for (int m = 8; m; m >>= 1) ss += __shfl_xor(ss, m);
    float r = rsqrtf(ss * (1.f / 64.f) + 1e-6f);
    #pragma unroll
    for (int j = 0; j < 4; ++j) ls[h][sub * 4 + j] = v[j] * r;
  }
  __syncthreads();
  #pragma unroll
  for (int j = 0; j < 4; ++j) {
    int e = sub * 4 + j;
    int i = e & 31;
    float invf = __powf(10000.f, -(float)i / 32.f);
    float ang = (float)s * invf;
    float sn, c; __sincosf(ang, &sn, &c);
    float x1 = ls[h][i], x2 = ls[h][i + 32];
    float o = (e < 32) ? (x1 * c + x2 * sn) : (x2 * c - x1 * sn);
    k_full[obase + 64 + e] = f32_bf16(o);
  }
  // v transpose
  #pragma unroll
  for (int j = 0; j < 8; ++j) {
    int d = sub * 8 + j;
    v_t[((size_t)(b * NH + h) * HDSZ + d) * SEQ + s] =
        f32_bf16(ur[NH * 64 + h * HDSZ + d]);
  }
}

// ---------------------------------------------------------------------------
// Causal flash attention, bf16 WMMA, f32 online softmax. Output y in bf16
// [B,S,H*HD] (K-contiguous operand of the final projection).
// ---------------------------------------------------------------------------
__global__ __launch_bounds__(256) void mla_attn(
    const unsigned short* __restrict__ q_full,
    const unsigned short* __restrict__ k_full,
    const unsigned short* __restrict__ v_t,
    unsigned short* __restrict__ y)
{
  const int bh = blockIdx.y;
  const int b = bh >> 4, h = bh & 15;
  const int lane = threadIdx.x & 31;
  const int wave = threadIdx.x >> 5;
  const int hi = lane >> 4;
  const int ln = lane & 15;
  const int q0 = blockIdx.x * 128 + wave * 16;

  const unsigned short* qf = q_full + (size_t)bh * SEQ * HDSZ;
  const unsigned short* kf = k_full + (size_t)bh * SEQ * HDSZ;
  const unsigned short* vt = v_t   + (size_t)bh * HDSZ * SEQ;

  __shared__ __align__(64) unsigned short p_lds[8][16 * 32];
  unsigned short* pw = p_lds[wave];

  bf16x16 aq[4];
  {
    const unsigned short* qrow = qf + (size_t)(q0 + ln) * HDSZ;
    #pragma unroll
    for (int g = 0; g < 4; ++g)
      aq[g] = frag_from2(qrow + g * 32 + hi * 16, qrow + g * 32 + hi * 16 + 8);
  }

  f32x8 acc[8];
  #pragma unroll
  for (int i = 0; i < 8; ++i) acc[i] = splat8(0.f);
  float mi[8], li[8];
  #pragma unroll
  for (int r = 0; r < 8; ++r) { mi[r] = -1e30f; li[r] = 0.f; }

  const float scale = 0.08838834764831845f; // 1/sqrt(128)

  for (int k0 = 0; k0 < q0 + 16; k0 += 32) {
    f32x8 sc[2];
    #pragma unroll
    for (int tt = 0; tt < 2; ++tt) {
      const int kt = k0 + tt * 16;
      if (kt > q0) { sc[tt] = splat8(-1e30f); continue; }
      f32x8 s = splat8(0.f);
      const unsigned short* krow = kf + (size_t)(kt + ln) * HDSZ;
      #pragma unroll
      for (int g = 0; g < 4; ++g) {
        bf16x16 bk = *(const bf16x16*)(krow + g * 32 + hi * 16);
        s = __builtin_amdgcn_wmma_f32_16x16x32_bf16(
            false, aq[g], false, bk, (short)0, s, false, false);
      }
      #pragma unroll
      for (int r = 0; r < 8; ++r) {
        float v = s[r] * scale;
        if (kt == q0 && ln > r + 8 * hi) v = -1e30f; // causal diagonal mask
        s[r] = v;
      }
      sc[tt] = s;
    }
    float mnew[8], fac[8];
    #pragma unroll
    for (int r = 0; r < 8; ++r) {
      float v = fmaxf(sc[0][r], sc[1][r]);
      #pragma unroll
      for (int m = 8; m; m >>= 1) v = fmaxf(v, __shfl_xor(v, m));
      mnew[r] = fmaxf(mi[r], v);
      fac[r] = __expf(mi[r] - mnew[r]);
      mi[r] = mnew[r];
    }
    #pragma unroll
    for (int tt = 0; tt < 2; ++tt)
      #pragma unroll
      for (int r = 0; r < 8; ++r) {
        float p = __expf(sc[tt][r] - mnew[r]);
        sc[tt][r] = p;
        pw[(r + 8 * hi) * 32 + tt * 16 + ln] = f32_bf16(p);
      }
    #pragma unroll
    for (int r = 0; r < 8; ++r) {
      float rs = sc[0][r] + sc[1][r];
      #pragma unroll
      for (int m = 8; m; m >>= 1) rs += __shfl_xor(rs, m);
      li[r] = li[r] * fac[r] + rs;
    }
    // intra-wave LDS store->load hazard (waves diverge: no __syncthreads here)
    asm volatile("s_wait_dscnt 0" ::: "memory");
    bf16x16 pa;
    {
      const unsigned short* pr = pw + ln * 32;
      pa = frag_from2(pr + hi * 16, pr + hi * 16 + 8);
    }
    #pragma unroll
    for (int nc = 0; nc < 8; ++nc)
      #pragma unroll
      for (int r = 0; r < 8; ++r) acc[nc][r] *= fac[r];
    #pragma unroll
    for (int nc = 0; nc < 8; ++nc) {
      const unsigned short* vr = vt + (size_t)(nc * 16 + ln) * SEQ + k0 + hi * 16;
      bf16x16 bv = *(const bf16x16*)vr;
      acc[nc] = __builtin_amdgcn_wmma_f32_16x16x32_bf16(
          false, pa, false, bv, (short)0, acc[nc], false, false);
    }
  }

  #pragma unroll
  for (int r = 0; r < 8; ++r) {
    const float inv = 1.f / li[r];
    const int qm = q0 + r + 8 * hi;
    unsigned short* yo = y + ((size_t)b * SEQ + qm) * DIMSZ + h * HDSZ;
    #pragma unroll
    for (int nc = 0; nc < 8; ++nc)
      yo[nc * 16 + ln] = f32_bf16(acc[nc][r] * inv);
  }
}

// ---------------------------------------------------------------------------
extern "C" void kernel_launch(void* const* d_in, const int* in_sizes, int n_in,
                              void* d_out, int out_size, void* d_ws, size_t ws_size,
                              hipStream_t stream)
{
  (void)in_sizes; (void)n_in; (void)out_size; (void)ws_size;
  const float* x         = (const float*)d_in[0];
  const float* w_q_krope = (const float*)d_in[1];
  const float* w_kv_down = (const float*)d_in[2];
  const float* w_kv_up   = (const float*)d_in[3];
  const float* w_proj    = (const float*)d_in[4];
  const float* q_gain    = (const float*)d_in[5];

  const int BS = NB * SEQ;                  // 4096
  char* ws = (char*)d_ws;
  size_t off = 0;
  auto alloc = [&](size_t bytes) { char* p = ws + off; off += (bytes + 255) & ~(size_t)255; return p; };

  unsigned short* xb      = (unsigned short*)alloc((size_t)BS * DIMSZ * 2);
  unsigned short* wqk_t   = (unsigned short*)alloc((size_t)W3 * DIMSZ * 2);
  unsigned short* wkvd_t  = (unsigned short*)alloc((size_t)LATD * DIMSZ * 2);
  unsigned short* wkvu_t  = (unsigned short*)alloc((size_t)W3 * LATD * 2);
  unsigned short* wproj_t = (unsigned short*)alloc((size_t)DIMSZ * DIMSZ * 2);
  float* q_krope          = (float*)alloc((size_t)BS * W3 * 4);
  unsigned short* y_b     = (unsigned short*)q_krope;  // alias: q_krope dead after prep
  float* ckv_raw          = (float*)alloc((size_t)BS * LATD * 4);
  unsigned short* ckv_b   = (unsigned short*)alloc((size_t)BS * LATD * 2);
  float* kv_up            = (float*)alloc((size_t)BS * W3 * 4);
  unsigned short* q_full  = (unsigned short*)alloc((size_t)NB * NH * SEQ * HDSZ * 2);
  unsigned short* k_full  = (unsigned short*)alloc((size_t)NB * NH * SEQ * HDSZ * 2);
  unsigned short* v_t     = (unsigned short*)alloc((size_t)NB * NH * SEQ * HDSZ * 2);

  dim3 blk(256);

  // pre-passes: one-time f32->bf16 (x) and transpose+convert (weights -> [N,K])
  cvt_f32_bf16<<<(BS * DIMSZ / 4 + 255) / 256, blk, 0, stream>>>(x, xb, BS * DIMSZ / 4);
  transpose_f32_bf16<<<dim3(W3 / 32, DIMSZ / 32), blk, 0, stream>>>(w_q_krope, wqk_t, DIMSZ, W3);
  transpose_f32_bf16<<<dim3(LATD / 32, DIMSZ / 32), blk, 0, stream>>>(w_kv_down, wkvd_t, DIMSZ, LATD);
  transpose_f32_bf16<<<dim3(W3 / 32, LATD / 32), blk, 0, stream>>>(w_kv_up, wkvu_t, LATD, W3);
  transpose_f32_bf16<<<dim3(DIMSZ / 32, DIMSZ / 32), blk, 0, stream>>>(w_proj, wproj_t, DIMSZ, DIMSZ);

  // 1) q_krope = x @ w_q_krope      [4096 x 3072]
  gemm_bf16_wmma<<<dim3(W3 / 128, BS / 128), blk, 0, stream>>>(
      xb, wqk_t, q_krope, BS, W3, DIMSZ);
  // 2) ckv_raw = x @ w_kv_down      [4096 x 512]
  gemm_bf16_wmma<<<dim3(LATD / 128, BS / 128), blk, 0, stream>>>(
      xb, wkvd_t, ckv_raw, BS, LATD, DIMSZ);
  // 3) ckv_b = rms_norm(ckv_raw) -> bf16
  rmsnorm_rows_bf16<<<BS, blk, 0, stream>>>(ckv_raw, ckv_b, LATD);
  // 4) kv_up = ckv_b @ w_kv_up      [4096 x 3072]
  gemm_bf16_wmma<<<dim3(W3 / 128, BS / 128), blk, 0, stream>>>(
      ckv_b, wkvu_t, kv_up, BS, W3, LATD);
  // 5) per-head norms + RoPE + gain -> bf16 q_full/k_full/v_t
  mla_prep<<<dim3(SEQ, NB), blk, 0, stream>>>(
      q_krope, kv_up, q_gain, q_full, k_full, v_t);
  // 6) causal flash attention -> y_b bf16 [B,S,2048] (aliases q_krope)
  mla_attn<<<dim3(SEQ / 128, NB * NH), blk, 0, stream>>>(
      q_full, k_full, v_t, y_b);
  // 7) out = y @ w_proj             [4096 x 2048]
  gemm_bf16_wmma<<<dim3(DIMSZ / 128, BS / 128), blk, 0, stream>>>(
      y_b, wproj_t, (float*)d_out, BS, DIMSZ, DIMSZ);
}
